// CorpusSimilarity_74526272520307
// MI455X (gfx1250) — matrix operand order repaired and checked by reading, not compiled, not run on hardware
//
#include <hip/hip_runtime.h>
#include <hip/hip_bf16.h>

typedef __attribute__((ext_vector_type(16))) __bf16 v16bf;
typedef __attribute__((ext_vector_type(8)))  __bf16 v8bf;
typedef __attribute__((ext_vector_type(8)))  float  v8f;

#define B_ROWS   256
#define IN_DIM   512
#define D_DIM    256
#define N_CORPUS 100000
#define NTILES   (N_CORPUS / 16)   // 6250
#define NWG      512
#define BROW     264               // 256 + 8 pad: 528B row stride (16B aligned, 4-bank rotate)
#define INV_2SIGD (1.0f / 512.0f)  // 1 / (D * 2 * sigma2)

// ---------------- Kernel 1: encoder  e = x@W + b  (fp32), emit bf16 e + fp32 ||e||^2 ----
__global__ void __launch_bounds__(256)
enc_kernel(const float* __restrict__ x, const float* __restrict__ W,
           const float* __restrict__ bias,
           __bf16* __restrict__ eB, float* __restrict__ e2) {
    const int i = blockIdx.x;    // row of e (0..255)
    const int j = threadIdx.x;   // col of e (0..255)
    const float* xr = x + i * IN_DIM;
    float s = bias[j];
#pragma unroll 8
    for (int k = 0; k < IN_DIM; ++k)
        s = fmaf(xr[k], W[k * D_DIM + j], s);
    eB[i * D_DIM + j] = (__bf16)s;

    __shared__ float red[256];
    red[j] = s * s;
    __syncthreads();
    for (int off = 128; off > 0; off >>= 1) {
        if (j < off) red[j] += red[j + off];
        __syncthreads();
    }
    if (j == 0) e2[i] = red[0];
}

// ---------------- Kernel 2: WMMA distance + exp accumulation --------------------------
// 256 threads = 8 waves; wave w owns M-tiles {2w, 2w+1} with A (bf16 e) resident in VGPRs.
// Corpus N-tile is loaded/converted ONCE per workgroup into double-buffered LDS (B layout
// order), with fp32 ||c||^2 computed during conversion. Compute waves read B via ds_load.
__global__ void __launch_bounds__(256)
corpus_kernel(const float* __restrict__ corpus,
              const __bf16* __restrict__ eB,
              const float*  __restrict__ e2,
              float* __restrict__ partials) {
    const int tid  = threadIdx.x;
    const int lane = tid & 31;
    const int wave = tid >> 5;            // 0..7
    const int l15  = lane & 15;
    const int hi   = lane >> 4;           // 0 or 1 (half-wave)
    const int wg   = blockIdx.x;

    __shared__ __bf16 Btile[2][16][BROW]; // 16.5 KB double-buffered bf16 corpus tile
    __shared__ float  c2s[2][16];         // fp32 ||c||^2 per tile row

    // staging slice for this thread: row (0..15), 16 consecutive cols
    const int srow = tid >> 4;            // tile row this thread converts
    const int scol = (tid & 15) * 16;     // k-base of its 16-element slice

    // ---- preload A tiles (ISA 16-bit A 16x32 layout) + e2 values ----
    v16bf A[2][8];
    float e2r[2][8];
#pragma unroll
    for (int t = 0; t < 2; ++t) {
        const int mt = 2 * wave + t;
        const int m  = mt * 16 + l15;
#pragma unroll
        for (int kc = 0; kc < 8; ++kc) {
            const __bf16* p = eB + m * D_DIM + kc * 32 + hi * 8;
            union { v16bf v; v8bf h[2]; } u;
            u.h[0] = *(const v8bf*)(p);        // K = kc*32 + hi*8 + {0..7}
            u.h[1] = *(const v8bf*)(p + 16);   // K = kc*32 + 16 + hi*8 + {0..7}
            A[t][kc] = u.v;
        }
#pragma unroll
        for (int r = 0; r < 8; ++r)
            e2r[t][r] = e2[mt * 16 + hi * 8 + r];
    }

    float part[2][8];
#pragma unroll
    for (int t = 0; t < 2; ++t)
#pragma unroll
        for (int r = 0; r < 8; ++r) part[t][r] = 0.0f;

    // ---- helper lambda: convert 16 fp32 -> bf16 slice into LDS buf + row ||c||^2 ----
    auto stage = [&](int buf, float4 f0, float4 f1, float4 f2, float4 f3) {
        float c2p = f0.x*f0.x + f0.y*f0.y + f0.z*f0.z + f0.w*f0.w
                  + f1.x*f1.x + f1.y*f1.y + f1.z*f1.z + f1.w*f1.w
                  + f2.x*f2.x + f2.y*f2.y + f2.z*f2.z + f2.w*f2.w
                  + f3.x*f3.x + f3.y*f3.y + f3.z*f3.z + f3.w*f3.w;
        v8bf b0, b1;
        b0[0]=(__bf16)f0.x; b0[1]=(__bf16)f0.y; b0[2]=(__bf16)f0.z; b0[3]=(__bf16)f0.w;
        b0[4]=(__bf16)f1.x; b0[5]=(__bf16)f1.y; b0[6]=(__bf16)f1.z; b0[7]=(__bf16)f1.w;
        b1[0]=(__bf16)f2.x; b1[1]=(__bf16)f2.y; b1[2]=(__bf16)f2.z; b1[3]=(__bf16)f2.w;
        b1[4]=(__bf16)f3.x; b1[5]=(__bf16)f3.y; b1[6]=(__bf16)f3.z; b1[7]=(__bf16)f3.w;
        *(v8bf*)(&Btile[buf][srow][scol])     = b0;   // 16B aligned (stride 528B)
        *(v8bf*)(&Btile[buf][srow][scol + 8]) = b1;
        // reduce c2 over the 16 threads of this row (one half-wave)
        c2p += __shfl_xor(c2p, 1, 32);
        c2p += __shfl_xor(c2p, 2, 32);
        c2p += __shfl_xor(c2p, 4, 32);
        c2p += __shfl_xor(c2p, 8, 32);
        if ((tid & 15) == 0) c2s[buf][srow] = c2p;
    };

    // ---- prologue: stage first tile ----
    int nt = wg;
    if (nt < NTILES) {
        const float4* cp = (const float4*)(corpus + (long)(nt * 16 + srow) * D_DIM + scol);
        stage(0, cp[0], cp[1], cp[2], cp[3]);
    }
    __syncthreads();

    int buf = 0;
    for (; nt < NTILES; nt += NWG) {
        const int nnext = nt + NWG;
        const bool have_next = (nnext < NTILES);
        // issue global prefetch of next tile early (hide latency under WMMA)
        float4 f0, f1, f2, f3;
        if (have_next) {
            const float4* cp = (const float4*)(corpus + (long)(nnext * 16 + srow) * D_DIM + scol);
            f0 = cp[0]; f1 = cp[1]; f2 = cp[2]; f3 = cp[3];
        }

        // ---- WMMA over K=256 reading B fragments from LDS ----
        v8f acc0 = {};
        v8f acc1 = {};
#pragma unroll
        for (int kc = 0; kc < 8; ++kc) {
            union { v16bf v; v8bf h[2]; } ub;
            const __bf16* bp = &Btile[buf][l15][kc * 32 + hi * 8];
            ub.h[0] = *(const v8bf*)(bp);       // local K = hi*8 + {0..7}
            ub.h[1] = *(const v8bf*)(bp + 16);  // local K = 16 + hi*8 + {0..7}
            acc0 = __builtin_amdgcn_wmma_f32_16x16x32_bf16(
                       false, A[0][kc], false, ub.v, (short)0, acc0, false, false);
            acc1 = __builtin_amdgcn_wmma_f32_16x16x32_bf16(
                       false, A[1][kc], false, ub.v, (short)0, acc1, false, false);
        }

        const float c2 = c2s[buf][l15];
#pragma unroll
        for (int r = 0; r < 8; ++r) {
            float d0 = fmaxf(e2r[0][r] + c2 - 2.0f * acc0[r], 0.0f);
            part[0][r] += __expf(d0 * (-INV_2SIGD));
            float d1 = fmaxf(e2r[1][r] + c2 - 2.0f * acc1[r], 0.0f);
            part[1][r] += __expf(d1 * (-INV_2SIGD));
        }

        // ---- stage next tile into the other buffer, then swap ----
        if (have_next) stage(buf ^ 1, f0, f1, f2, f3);
        __syncthreads();
        buf ^= 1;
    }

    // reduce over the 16 lanes that share the same m (halves stay separate: masks < 16)
#pragma unroll
    for (int t = 0; t < 2; ++t) {
        const int mt = 2 * wave + t;
#pragma unroll
        for (int r = 0; r < 8; ++r) {
            float v = part[t][r];
            v += __shfl_xor(v, 1, 32);
            v += __shfl_xor(v, 2, 32);
            v += __shfl_xor(v, 4, 32);
            v += __shfl_xor(v, 8, 32);
            if (l15 == 0)   // lanes 0 and 16 hold the two halves' sums (m differs by 8)
                partials[wg * 256 + mt * 16 + hi * 8 + r] = v;
        }
    }
}

// ---------------- Kernel 3: reduce per-WG partials, scale by 1/N ----------------------
__global__ void __launch_bounds__(256)
reduce_kernel(const float* __restrict__ partials, float* __restrict__ out) {
    const int m = threadIdx.x;
    float s = 0.0f;
    for (int g = 0; g < NWG; ++g) s += partials[g * 256 + m];
    out[m] = s * (1.0f / (float)N_CORPUS);
}

extern "C" void kernel_launch(void* const* d_in, const int* in_sizes, int n_in,
                              void* d_out, int out_size, void* d_ws, size_t ws_size,
                              hipStream_t stream) {
    (void)in_sizes; (void)n_in; (void)out_size; (void)ws_size;
    const float* x      = (const float*)d_in[0];  // explicand [256,512]
    const float* W      = (const float*)d_in[1];  // [512,256]
    const float* bias   = (const float*)d_in[2];  // [256]
    const float* corpus = (const float*)d_in[3];  // [100000,256]
    float* out = (float*)d_out;                   // [256]

    char* ws = (char*)d_ws;
    __bf16* eB       = (__bf16*)(ws);                   // 256*256*2 = 131072 B
    float*  e2       = (float*)(ws + 131072);           // 256*4     =   1024 B
    float*  partials = (float*)(ws + 131072 + 1024);    // NWG*256*4 = 524288 B

    enc_kernel<<<256, 256, 0, stream>>>(x, W, bias, eB, e2);
    corpus_kernel<<<NWG, 256, 0, stream>>>(corpus, eB, e2, partials);
    reduce_kernel<<<1, 256, 0, stream>>>(partials, out);
}
